// Transformer_68736656605718
// MI455X (gfx1250) — matrix operand order repaired and checked by reading, not compiled, not run on hardware
//
#include <hip/hip_runtime.h>
#include <cmath>

// ---------------------------------------------------------------------------
// CDNA5 (gfx1250) fused conv-attention transformer block.
// - bf16 WMMA (v_wmma_f32_16x16x32_bf16) everywhere, f32 accumulation.
// - LDS tiles in WMMA fragment-major order: each fragment load is one
//   contiguous 32B/lane read (2x ds_load_b128, no repacking).
// - Tile fills use CDNA5 async global->LDS (global_load_async_to_lds_b128,
//   ASYNCcnt) from pre-packed fragment-order buffers.
// ---------------------------------------------------------------------------

typedef __bf16 bf16_t;
typedef __attribute__((ext_vector_type(16))) __bf16 v16bf;
typedef __attribute__((ext_vector_type(8)))  __bf16 v8bf;
typedef __attribute__((ext_vector_type(8)))  float  v8f;

constexpr int   BATCH = 8;
constexpr int   DIMC  = 128;
constexpr int   HEADS = 8;
constexpr int   SEQL  = 4096;
constexpr int   INNER = DIMC * HEADS;   // 1024
constexpr int   NPOS  = BATCH * SEQL;   // 32768
constexpr int   FFD   = 4 * DIMC;       // 512
constexpr float EPS_  = 1e-5f;
constexpr float SCALE_ = 0.08838834764831845f;  // 128^-0.5

// Block tile 128x128, K-step 32, 256 threads (8 waves as 2x4),
// wave tile 64x32 = 4x2 WMMA 16x16 accumulators.
constexpr int BM = 128, BN = 128, BK = 32;
constexpr int TILE_ELEMS = BM * BK;     // 4096 per tile (A and B)

// ---- fragment-major indexing (layouts per CDNA5 ISA) -----------
// A 16x32 frag: lane = half*16 + (m&15); half=(k>>3)&1, e=(k&7)+8*((k>>4)&1)
__device__ __host__ inline int a_dst(int m, int k) {
  const int sub  = m >> 4;
  const int half = (k >> 3) & 1;
  const int e    = (k & 7) + ((k >> 4) & 1) * 8;
  return ((sub * 32 + half * 16 + (m & 15)) << 4) + e;
}
// B 32x16 frag: lane = half*16 + (n&15); half=(k>>3)&1, e=(k&7)+8*((k>>4)&1)
__device__ __host__ inline int b_dst(int k, int n) {
  const int sub  = n >> 4;
  const int half = (k >> 3) & 1;
  return ((sub * 32 + half * 16 + (n & 15)) << 4) + (k & 7) + ((k >> 4) & 1) * 8;
}
__device__ inline v16bf frag_ld(const bf16_t* lds, int sub, int lane) {
  return *(const v16bf*)(lds + ((sub * 32 + lane) << 4));
}
__device__ inline v8f wmma_bf16(v16bf a, v16bf b, v8f c) {
  return __builtin_amdgcn_wmma_f32_16x16x32_bf16(false, a, false, b,
                                                 (short)0, c, false, false);
}

// ---- CDNA5 async global->LDS copy (16B per lane), tracked by ASYNCcnt -----
__device__ inline void async_cp16(bf16_t* lds_dst, const bf16_t* gsrc) {
  const unsigned laddr = (unsigned)(size_t)lds_dst;  // low 32b = LDS offset
  const unsigned long long gaddr = (unsigned long long)(size_t)gsrc;
  asm volatile("global_load_async_to_lds_b128 %0, %1, off"
               :: "v"(laddr), "v"(gaddr) : "memory");
}
__device__ inline void async_wait0() {
  asm volatile("s_wait_asynccnt 0x0" ::: "memory");
}

// ---------------------------------------------------------------------------
// Generic GEMM: C[M,N] = A[M,K] (bf16) * Wp (bf16, fragment-packed tiles)
// APACK=1: A is fragment-packed tiles too (qkv path); else row-major (lda).
// MODE 0: store bf16              MODE 1: sigmoid -> bf16
// MODE 2: exact GELU -> bf16      MODE 3: +bias +x(NCL residual) -> f32 [pos,128]
// MODE 4: +bias +res[pos,128] -> f32 transposed store [B,DIM,L]
// ---------------------------------------------------------------------------
template <int MODE, bool APACK>
__global__ __launch_bounds__(256) void gemm_kernel(
    const bf16_t* __restrict__ A, int lda,
    const bf16_t* __restrict__ Wp, const float* __restrict__ bias,
    int Ktot,
    bf16_t* __restrict__ outb, int ldc,
    const float* __restrict__ resx,    // MODE 3: x in [B,DIM,L]
    const float* __restrict__ resrow,  // MODE 4: residual [pos,128]
    float* __restrict__ outf)          // MODE 3/4 f32 destination
{
  __shared__ __align__(32) bf16_t lA[TILE_ELEMS];
  __shared__ __align__(32) bf16_t lB[TILE_ELEMS];

  const int tid  = threadIdx.x;
  const int lane = tid & 31;
  const int wid  = tid >> 5;
  const int wm   = (wid & 1) * 64;
  const int wn   = (wid >> 1) * 32;
  const int m0   = blockIdx.x * BM;
  const int n0   = blockIdx.y * BN;
  const int ktiles = Ktot >> 5;

  v8f acc[4][2];
#pragma unroll
  for (int i = 0; i < 4; ++i)
#pragma unroll
    for (int j = 0; j < 2; ++j) acc[i][j] = v8f{};

  const long abase = APACK ? (long)(m0 >> 7) * ktiles * TILE_ELEMS
                           : (long)m0 * lda;
  const long wbase = (long)(n0 >> 7) * ktiles * TILE_ELEMS;

  for (int k0 = 0; k0 < Ktot; k0 += BK) {
    // ---- A tile fill: async 16B copies ----
    if (APACK) {
      const bf16_t* src = A + abase + (long)(k0 >> 5) * TILE_ELEMS + tid * 16;
      async_cp16(lA + tid * 16, src);
      async_cp16(lA + tid * 16 + 8, src + 8);
    } else {
#pragma unroll
      for (int i = 0; i < 2; ++i) {
        const int c = tid + i * 256;
        const int k8 = (c & 3) * 8, m = c >> 2;
        async_cp16(lA + a_dst(m, k8), A + abase + (long)m * lda + (k0 + k8));
      }
    }
    // ---- B tile fill: packed weights, contiguous async ----
    {
      const bf16_t* src = Wp + wbase + (long)(k0 >> 5) * TILE_ELEMS + tid * 16;
      async_cp16(lB + tid * 16, src);
      async_cp16(lB + tid * 16 + 8, src + 8);
    }
    async_wait0();
    __syncthreads();

    v16bf bfr[2];
#pragma unroll
    for (int ni = 0; ni < 2; ++ni) bfr[ni] = frag_ld(lB, (wn >> 4) + ni, lane);
#pragma unroll
    for (int mi = 0; mi < 4; ++mi) {
      const v16bf af = frag_ld(lA, (wm >> 4) + mi, lane);
#pragma unroll
      for (int ni = 0; ni < 2; ++ni) acc[mi][ni] = wmma_bf16(af, bfr[ni], acc[mi][ni]);
    }
    __syncthreads();
  }

  // epilogue (C layout: VGPR r -> row r + 8*(lane>=16), col = lane&15)
#pragma unroll
  for (int mi = 0; mi < 4; ++mi)
#pragma unroll
    for (int ni = 0; ni < 2; ++ni) {
      const int col   = n0 + wn + ni * 16 + (lane & 15);
      const int rbase = m0 + wm + mi * 16 + (lane >> 4) * 8;
#pragma unroll
      for (int r = 0; r < 8; ++r) {
        const int row = rbase + r;
        float v = acc[mi][ni][r];
        if (bias) v += bias[col];
        if (MODE == 0) {
          outb[(long)row * ldc + col] = (bf16_t)v;
        } else if (MODE == 1) {
          outb[(long)row * ldc + col] = (bf16_t)(1.f / (1.f + expf(-v)));
        } else if (MODE == 2) {
          outb[(long)row * ldc + col] = (bf16_t)(0.5f * v * (1.f + erff(v * 0.70710678118f)));
        } else if (MODE == 3) {
          const int b = row / SEQL, l = row % SEQL;
          outf[(long)row * DIMC + col] = v + resx[((long)b * DIMC + col) * SEQL + l];
        } else if (MODE == 4) {
          const int b = row / SEQL, l = row % SEQL;
          outf[((long)b * DIMC + col) * SEQL + l] = v + resrow[(long)row * DIMC + col];
        }
      }
    }
}

// ---------------------------------------------------------------------------
// kk[n] += q[n]^T * ksoft[n]  (128x128, K=4096 split 8 ways, f32 atomics)
// Both operands transposed -> VGPR scatter fill.
// ---------------------------------------------------------------------------
__global__ __launch_bounds__(256) void kk_gemm_kernel(
    const bf16_t* __restrict__ qb, const bf16_t* __restrict__ ks,
    float* __restrict__ kkf)
{
  __shared__ __align__(32) bf16_t lA[TILE_ELEMS];
  __shared__ __align__(32) bf16_t lB[TILE_ELEMS];
  const int tid = threadIdx.x, lane = tid & 31, wid = tid >> 5;
  const int wm = (wid & 1) * 64, wn = (wid >> 1) * 32;
  const int n = blockIdx.y, kc = blockIdx.x;
  const int b = n >> 3, h = n & 7;
  const long base = (long)b * SEQL * INNER + h * DIMC;

  v8f acc[4][2];
#pragma unroll
  for (int i = 0; i < 4; ++i)
#pragma unroll
    for (int j = 0; j < 2; ++j) acc[i][j] = v8f{};

  for (int x0 = kc * 512; x0 < kc * 512 + 512; x0 += BK) {
#pragma unroll
    for (int i = 0; i < 2; ++i) {
      const int c = tid + i * 256;
      const int d8 = (c & 15) * 8, k = c >> 4;
      const long src = base + (long)(x0 + k) * INNER + d8;
      const v8bf av = *(const v8bf*)(qb + src);  // A(m=d, k=x)
      const v8bf bv = *(const v8bf*)(ks + src);  // B(k=x, n=e)
#pragma unroll
      for (int j = 0; j < 8; ++j) {
        lA[a_dst(d8 + j, k)] = av[j];
        lB[b_dst(k, d8 + j)] = bv[j];
      }
    }
    __syncthreads();
    v16bf bfr[2];
#pragma unroll
    for (int ni = 0; ni < 2; ++ni) bfr[ni] = frag_ld(lB, (wn >> 4) + ni, lane);
#pragma unroll
    for (int mi = 0; mi < 4; ++mi) {
      const v16bf af = frag_ld(lA, (wm >> 4) + mi, lane);
#pragma unroll
      for (int ni = 0; ni < 2; ++ni) acc[mi][ni] = wmma_bf16(af, bfr[ni], acc[mi][ni]);
    }
    __syncthreads();
  }
#pragma unroll
  for (int mi = 0; mi < 4; ++mi)
#pragma unroll
    for (int ni = 0; ni < 2; ++ni) {
      const int col = wn + ni * 16 + (lane & 15);
      const int rb  = wm + mi * 16 + (lane >> 4) * 8;
#pragma unroll
      for (int r = 0; r < 8; ++r)
        unsafeAtomicAdd(&kkf[((long)n * 128 + rb + r) * 128 + col], acc[mi][ni][r]);
    }
}

// kk f32 -> bf16 *SCALE, written fragment-packed (B-tiles of 32 k-rows)
__global__ __launch_bounds__(256) void kkcvt_kernel(const float* __restrict__ kkf,
                                                    bf16_t* __restrict__ kkb) {
  const long i = (long)blockIdx.x * 256 + threadIdx.x;  // over 64*128*128
  const int e = (int)(i & 127);
  const int d = (int)((i >> 7) & 127);
  const int n = (int)(i >> 14);
  kkb[(long)n * 16384 + (long)(d >> 5) * TILE_ELEMS + b_dst(d & 31, e)] =
      (bf16_t)(kkf[i] * SCALE_);
}

// ---------------------------------------------------------------------------
// out = v @ kk[n] + gate*v  (gate buffer holds sigmoid(g); updated in place)
// A row-major (async), B fragment-packed kk (async).
// ---------------------------------------------------------------------------
__global__ __launch_bounds__(256) void vkk_gemm_kernel(
    const bf16_t* __restrict__ vb, const bf16_t* __restrict__ kkb,
    bf16_t* __restrict__ gio)
{
  __shared__ __align__(32) bf16_t lA[TILE_ELEMS];
  __shared__ __align__(32) bf16_t lB[TILE_ELEMS];
  const int tid = threadIdx.x, lane = tid & 31, wid = tid >> 5;
  const int wm = (wid & 1) * 64, wn = (wid >> 1) * 32;
  const int n = blockIdx.y, x0 = blockIdx.x * BM;
  const int b = n >> 3, h = n & 7;
  const long base = ((long)b * SEQL + x0) * INNER + h * DIMC;
  const long kkbase = (long)n * 16384;

  v8f acc[4][2];
#pragma unroll
  for (int i = 0; i < 4; ++i)
#pragma unroll
    for (int j = 0; j < 2; ++j) acc[i][j] = v8f{};

  for (int k0 = 0; k0 < DIMC; k0 += BK) {
#pragma unroll
    for (int i = 0; i < 2; ++i) {
      const int c = tid + i * 256;
      const int k8 = (c & 3) * 8, m = c >> 2;
      async_cp16(lA + a_dst(m, k8), vb + base + (long)m * INNER + (k0 + k8));
    }
    {
      const bf16_t* src = kkb + kkbase + (long)(k0 >> 5) * TILE_ELEMS + tid * 16;
      async_cp16(lB + tid * 16, src);
      async_cp16(lB + tid * 16 + 8, src + 8);
    }
    async_wait0();
    __syncthreads();
    v16bf bfr[2];
#pragma unroll
    for (int ni = 0; ni < 2; ++ni) bfr[ni] = frag_ld(lB, (wn >> 4) + ni, lane);
#pragma unroll
    for (int mi = 0; mi < 4; ++mi) {
      const v16bf af = frag_ld(lA, (wm >> 4) + mi, lane);
#pragma unroll
      for (int ni = 0; ni < 2; ++ni) acc[mi][ni] = wmma_bf16(af, bfr[ni], acc[mi][ni]);
    }
    __syncthreads();
  }
#pragma unroll
  for (int mi = 0; mi < 4; ++mi)
#pragma unroll
    for (int ni = 0; ni < 2; ++ni) {
      const int col = wn + ni * 16 + (lane & 15);
      const int rb  = wm + mi * 16 + (lane >> 4) * 8;
#pragma unroll
      for (int r = 0; r < 8; ++r) {
        const long off = base + (long)(rb + r) * INNER + col;
        const float g = (float)gio[off];
        const float v = (float)vb[off];
        gio[off] = (bf16_t)(acc[mi][ni][r] + g * v);
      }
    }
}

// ---------------------------------------------------------------------------
// pack f32 weight [N,K] row-major -> bf16 fragment-packed tiles
// tile id = (n>>7)*(K>>5) + (k>>5); within-tile: b_dst(k&31, n&127)
// ---------------------------------------------------------------------------
__global__ __launch_bounds__(256) void pack_w_kernel(const float* __restrict__ W,
                                                     bf16_t* __restrict__ P,
                                                     int Ktot) {
  const long i = (long)blockIdx.x * 256 + threadIdx.x;
  const int k = (int)(i % Ktot);
  const int n = (int)(i / Ktot);
  const long tile = (long)(n >> 7) * (Ktot >> 5) + (k >> 5);
  P[tile * TILE_ELEMS + b_dst(k & 31, n & 127)] = (bf16_t)W[i];
}

// ---------------------------------------------------------------------------
// LayerNorm over channels, x in [B,DIM,L]; one wave per position.
// ---------------------------------------------------------------------------
__global__ __launch_bounds__(256) void ln1_kernel(const float* __restrict__ x,
                                                  const float* __restrict__ g,
                                                  const float* __restrict__ bt,
                                                  float* __restrict__ xn) {
  const int w = blockIdx.x * 8 + (threadIdx.x >> 5);
  const int lane = threadIdx.x & 31;
  const int b = w / SEQL, l = w % SEQL;
  const long base = (long)b * DIMC * SEQL + l;
  float v[4], s = 0.f, sq = 0.f;
#pragma unroll
  for (int j = 0; j < 4; ++j) {
    const int c = lane + 32 * j;
    v[j] = x[base + (long)c * SEQL];
    s += v[j]; sq += v[j] * v[j];
  }
  for (int o = 16; o > 0; o >>= 1) { s += __shfl_xor(s, o, 32); sq += __shfl_xor(sq, o, 32); }
  const float mean = s * (1.f / 128.f);
  const float rstd = rsqrtf(sq * (1.f / 128.f) - mean * mean + EPS_);
#pragma unroll
  for (int j = 0; j < 4; ++j) {
    const int c = lane + 32 * j;
    xn[base + (long)c * SEQL] = (v[j] - mean) * rstd * g[c] + bt[c];
  }
}

// LayerNorm over row-major [pos,128] residual -> bf16 [pos,128]
__global__ __launch_bounds__(256) void ln2_kernel(const float* __restrict__ xr,
                                                  const float* __restrict__ g,
                                                  const float* __restrict__ bt,
                                                  bf16_t* __restrict__ xn2) {
  const int w = blockIdx.x * 8 + (threadIdx.x >> 5);
  const int lane = threadIdx.x & 31;
  const long base = (long)w * DIMC;
  float v[4], s = 0.f, sq = 0.f;
#pragma unroll
  for (int j = 0; j < 4; ++j) {
    const int c = lane + 32 * j;
    v[j] = xr[base + c];
    s += v[j]; sq += v[j] * v[j];
  }
  for (int o = 16; o > 0; o >>= 1) { s += __shfl_xor(s, o, 32); sq += __shfl_xor(sq, o, 32); }
  const float mean = s * (1.f / 128.f);
  const float rstd = rsqrtf(sq * (1.f / 128.f) - mean * mean + EPS_);
#pragma unroll
  for (int j = 0; j < 4; ++j) {
    const int c = lane + 32 * j;
    xn2[base + c] = (bf16_t)((v[j] - mean) * rstd * g[c] + bt[c]);
  }
}

// depthwise conv3 + folded BN; reads xn [B,DIM,L] coalesced, writes output
// directly in A-fragment-packed order for the qkv GEMMs.
__global__ __launch_bounds__(256) void dwbn_kernel(
    const float* __restrict__ xn, const float* __restrict__ dw,
    const float* __restrict__ bg, const float* __restrict__ bb,
    const float* __restrict__ bm, const float* __restrict__ bv,
    bf16_t* __restrict__ yp) {
  const long gid = (long)blockIdx.x * 256 + threadIdx.x;
  const int l = (int)(gid % SEQL);
  const long t = gid / SEQL;
  const int c = (int)(t % DIMC);
  const int b = (int)(t / DIMC);
  const long base = t * SEQL;
  const float s0 = (l > 0) ? xn[base + l - 1] : 0.f;
  const float s1 = xn[base + l];
  const float s2 = (l < SEQL - 1) ? xn[base + l + 1] : 0.f;
  const float conv = dw[c * 3 + 0] * s0 + dw[c * 3 + 1] * s1 + dw[c * 3 + 2] * s2;
  const float sc = bg[c] * rsqrtf(bv[c] + EPS_);
  const int pos = b * SEQL + l;
  const long tile = (long)(pos >> 7) * (DIMC >> 5) + (c >> 5);
  yp[tile * TILE_ELEMS + a_dst(pos & 127, c & 31)] =
      (bf16_t)(conv * sc + (bb[c] - bm[c] * sc));
}

// softmax over each 128-slice of [pos*HEADS, 128] (in place, bf16)
__global__ __launch_bounds__(256) void softmax_kernel(bf16_t* __restrict__ kb) {
  const long s = (long)blockIdx.x * 8 + (threadIdx.x >> 5);
  const int lane = threadIdx.x & 31;
  const long base = s * 128;
  float v[4];
#pragma unroll
  for (int j = 0; j < 4; ++j) v[j] = (float)kb[base + lane + 32 * j];
  float m = fmaxf(fmaxf(v[0], v[1]), fmaxf(v[2], v[3]));
  for (int o = 16; o > 0; o >>= 1) m = fmaxf(m, __shfl_xor(m, o, 32));
  float sum = 0.f;
#pragma unroll
  for (int j = 0; j < 4; ++j) { v[j] = expf(v[j] - m); sum += v[j]; }
  for (int o = 16; o > 0; o >>= 1) sum += __shfl_xor(sum, o, 32);
  const float inv = 1.f / sum;
#pragma unroll
  for (int j = 0; j < 4; ++j) kb[base + lane + 32 * j] = (bf16_t)(v[j] * inv);
}

// fold dwc (kernel-1 depthwise + BN) into the 1x1 weight: W'[o,d], bias'[o]
__global__ void wgate_kernel(const float* __restrict__ pw, const float* __restrict__ dwd,
                             const float* __restrict__ bg, const float* __restrict__ bb,
                             const float* __restrict__ bm, const float* __restrict__ bv,
                             float* __restrict__ wg, float* __restrict__ bgate) {
  const int o = threadIdx.x;  // 0..127
  float acc = 0.f;
  for (int d = 0; d < DIMC; ++d) {
    const float sd = bg[d] * rsqrtf(bv[d] + EPS_);
    const float p = pw[o * DIMC + d];
    wg[o * DIMC + d] = p * dwd[d] * sd;
    acc += p * (bb[d] - bm[d] * sd);
  }
  bgate[o] = acc;
}

// ---------------------------------------------------------------------------
extern "C" void kernel_launch(void* const* d_in, const int* in_sizes, int n_in,
                              void* d_out, int out_size, void* d_ws, size_t ws_size,
                              hipStream_t stream) {
  const float* x      = (const float*)d_in[0];
  const float* ln1_g  = (const float*)d_in[1];
  const float* ln1_b  = (const float*)d_in[2];
  const float* q_dw   = (const float*)d_in[3];
  const float* q_bng  = (const float*)d_in[4];
  const float* q_bnb  = (const float*)d_in[5];
  const float* q_bnm  = (const float*)d_in[6];
  const float* q_bnv  = (const float*)d_in[7];
  const float* q_pw   = (const float*)d_in[8];
  const float* k_dw   = (const float*)d_in[9];
  const float* k_bng  = (const float*)d_in[10];
  const float* k_bnb  = (const float*)d_in[11];
  const float* k_bnm  = (const float*)d_in[12];
  const float* k_bnv  = (const float*)d_in[13];
  const float* k_pw   = (const float*)d_in[14];
  const float* v_dw   = (const float*)d_in[15];
  const float* v_bng  = (const float*)d_in[16];
  const float* v_bnb  = (const float*)d_in[17];
  const float* v_bnm  = (const float*)d_in[18];
  const float* v_bnv  = (const float*)d_in[19];
  const float* v_pw   = (const float*)d_in[20];
  const float* dwc_dw = (const float*)d_in[21];
  const float* dwc_bg = (const float*)d_in[22];
  const float* dwc_bb = (const float*)d_in[23];
  const float* dwc_bm = (const float*)d_in[24];
  const float* dwc_bv = (const float*)d_in[25];
  const float* dwc_pw = (const float*)d_in[26];
  const float* out_w  = (const float*)d_in[27];
  const float* out_b  = (const float*)d_in[28];
  const float* ln2_g  = (const float*)d_in[29];
  const float* ln2_b  = (const float*)d_in[30];
  const float* ff_w1  = (const float*)d_in[31];
  const float* ff_b1  = (const float*)d_in[32];
  const float* ff_w2  = (const float*)d_in[33];
  const float* ff_b2  = (const float*)d_in[34];
  float* out = (float*)d_out;

  char* ws = (char*)d_ws;
  size_t off = 0;
  auto take = [&](size_t bytes) { char* p = ws + off; off += (bytes + 255) & ~(size_t)255; return p; };

  float*  xn    = (float*)take((size_t)NPOS * DIMC * 4);       // [B,DIM,L]
  bf16_t* ypq   = (bf16_t*)take((size_t)NPOS * DIMC * 2);      // A-frag packed
  bf16_t* ypk   = (bf16_t*)take((size_t)NPOS * DIMC * 2);
  bf16_t* ypv   = (bf16_t*)take((size_t)NPOS * DIMC * 2);
  bf16_t* qb    = (bf16_t*)take((size_t)NPOS * INNER * 2);     // [pos,1024]
  bf16_t* kb    = (bf16_t*)take((size_t)NPOS * INNER * 2);     // softmax in place
  bf16_t* vb    = (bf16_t*)take((size_t)NPOS * INNER * 2);
  float*  kkf   = (float*)take((size_t)64 * 128 * 128 * 4);
  bf16_t* kkb   = (bf16_t*)take((size_t)64 * 128 * 128 * 2);   // B-frag packed
  float*  wg    = (float*)take((size_t)DIMC * DIMC * 4);
  float*  bg    = (float*)take((size_t)DIMC * 4);
  bf16_t* gb    = (bf16_t*)take((size_t)NPOS * INNER * 2);     // gate -> attn-out
  float*  xres  = (float*)take((size_t)NPOS * DIMC * 4);       // [pos,128]
  bf16_t* xn2   = (bf16_t*)take((size_t)NPOS * DIMC * 2);
  bf16_t* hb    = (bf16_t*)take((size_t)NPOS * FFD * 2);
  bf16_t* qpw_p = (bf16_t*)take((size_t)INNER * DIMC * 2);     // packed weights
  bf16_t* kpw_p = (bf16_t*)take((size_t)INNER * DIMC * 2);
  bf16_t* vpw_p = (bf16_t*)take((size_t)INNER * DIMC * 2);
  bf16_t* wg_p  = (bf16_t*)take((size_t)DIMC * DIMC * 2);
  bf16_t* ow_p  = (bf16_t*)take((size_t)DIMC * INNER * 2);
  bf16_t* f1_p  = (bf16_t*)take((size_t)FFD * DIMC * 2);
  bf16_t* f2_p  = (bf16_t*)take((size_t)DIMC * FFD * 2);
  (void)ws_size; (void)n_in; (void)in_sizes; (void)out_size;

  // 0) pack all static weights into fragment-major tile order (bf16)
  pack_w_kernel<<<(INNER * DIMC) / 256, 256, 0, stream>>>(q_pw, qpw_p, DIMC);
  pack_w_kernel<<<(INNER * DIMC) / 256, 256, 0, stream>>>(k_pw, kpw_p, DIMC);
  pack_w_kernel<<<(INNER * DIMC) / 256, 256, 0, stream>>>(v_pw, vpw_p, DIMC);
  pack_w_kernel<<<(DIMC * INNER) / 256, 256, 0, stream>>>(out_w, ow_p, INNER);
  pack_w_kernel<<<(FFD * DIMC) / 256, 256, 0, stream>>>(ff_w1, f1_p, DIMC);
  pack_w_kernel<<<(DIMC * FFD) / 256, 256, 0, stream>>>(ff_w2, f2_p, FFD);
  // 1) LN1
  ln1_kernel<<<NPOS / 8, 256, 0, stream>>>(x, ln1_g, ln1_b, xn);
  // 2) depthwise conv3 + folded BN -> A-fragment-packed bf16
  const int dwblocks = (NPOS * DIMC) / 256;
  dwbn_kernel<<<dwblocks, 256, 0, stream>>>(xn, q_dw, q_bng, q_bnb, q_bnm, q_bnv, ypq);
  dwbn_kernel<<<dwblocks, 256, 0, stream>>>(xn, k_dw, k_bng, k_bnb, k_bnm, k_bnv, ypk);
  dwbn_kernel<<<dwblocks, 256, 0, stream>>>(xn, v_dw, v_bng, v_bnb, v_bnm, v_bnv, ypv);
  // 3) 1x1 projections (M=32768, N=1024, K=128), A + B both async/packed
  gemm_kernel<0, true><<<dim3(NPOS / BM, INNER / BN), 256, 0, stream>>>(
      ypq, 0, qpw_p, nullptr, DIMC, qb, INNER, nullptr, nullptr, nullptr);
  gemm_kernel<0, true><<<dim3(NPOS / BM, INNER / BN), 256, 0, stream>>>(
      ypk, 0, kpw_p, nullptr, DIMC, kb, INNER, nullptr, nullptr, nullptr);
  gemm_kernel<0, true><<<dim3(NPOS / BM, INNER / BN), 256, 0, stream>>>(
      ypv, 0, vpw_p, nullptr, DIMC, vb, INNER, nullptr, nullptr, nullptr);
  // 4) softmax over feature dim per (pos, head)
  softmax_kernel<<<(NPOS * HEADS) / 8, 256, 0, stream>>>(kb);
  // 5) gate: fold dwc affine into weight, pack, GEMM + sigmoid epilogue
  wgate_kernel<<<1, DIMC, 0, stream>>>(dwc_pw, dwc_dw, dwc_bg, dwc_bb, dwc_bm, dwc_bv, wg, bg);
  pack_w_kernel<<<(DIMC * DIMC) / 256, 256, 0, stream>>>(wg, wg_p, DIMC);
  gemm_kernel<1, false><<<dim3((NPOS * HEADS) / BM, 1), 256, 0, stream>>>(
      qb, DIMC, wg_p, bg, DIMC, gb, DIMC, nullptr, nullptr, nullptr);
  // 6) kk = q^T @ softmax(k), K=4096 split 8 ways with f32 atomics
  hipMemsetAsync(kkf, 0, (size_t)64 * 128 * 128 * 4, stream);
  kk_gemm_kernel<<<dim3(8, 64), 256, 0, stream>>>(qb, kb, kkf);
  kkcvt_kernel<<<(64 * 128 * 128) / 256, 256, 0, stream>>>(kkf, kkb);
  // 7) out = v @ kk + sigmoid(g) * v  (in place over gate buffer)
  vkk_gemm_kernel<<<dim3(SEQL / BM, 64), 256, 0, stream>>>(vb, kkb, gb);
  // 8) output projection + bias + residual(x) -> xres [pos,128] f32
  gemm_kernel<3, false><<<dim3(NPOS / BM, 1), 256, 0, stream>>>(
      gb, INNER, ow_p, out_b, INNER, nullptr, 0, x, nullptr, xres);
  // 9) LN2 -> bf16
  ln2_kernel<<<NPOS / 8, 256, 0, stream>>>(xres, ln2_g, ln2_b, xn2);
  // 10) FFN: 128->512 (+bias, exact GELU), 512->128 (+bias + residual)
  gemm_kernel<2, false><<<dim3(NPOS / BM, FFD / BN), 256, 0, stream>>>(
      xn2, DIMC, f1_p, ff_b1, DIMC, hb, FFD, nullptr, nullptr, nullptr);
  gemm_kernel<4, false><<<dim3(NPOS / BM, 1), 256, 0, stream>>>(
      hb, FFD, f2_p, ff_b2, FFD, nullptr, 0, nullptr, xres, out);
}
